// Model_9603546874263
// MI455X (gfx1250) — matrix operand order repaired and checked by reading, not compiled, not run on hardware
//
#include <hip/hip_runtime.h>

// ---------------------------------------------------------------------------
// Seq2seq GRU (BiGRU encoder + GRU decoder + vocab projection) for MI455X.
// Compute-bound (~600 GFLOP) in sequential [4096,512]x[512,1536] GEMMs ->
// bf16 WMMA (v_wmma_f32_16x16x32_bf16) with f32 accumulation, fused gate
// epilogue per step. Weights pre-transposed to column-major bf16 so B
// fragments are single contiguous 32B loads per lane.
// ---------------------------------------------------------------------------

#define B_   4096
#define S_   24
#define T_   24
#define E_   128
#define H_   512
#define VT_  256
#define G3_  1536   // 3*H
#define KC_  640    // H + E (fused [h | x] K dimension)

typedef __attribute__((ext_vector_type(16))) __bf16 v16bf;
typedef __attribute__((ext_vector_type(8)))  __bf16 v8bf;
typedef __attribute__((ext_vector_type(8)))  float  v8f;

__device__ __forceinline__ v8f wmma_bf16(v16bf a, v16bf b, v8f c) {
    // (neg_a, A, neg_b, B, c_mod, C, reuse_a, reuse_b)
    return __builtin_amdgcn_wmma_f32_16x16x32_bf16(false, a, false, b,
                                                   (short)0, c, false, false);
}

// A fragment: 16x32 bf16, M = lane%16, lanes 0-15 hold K {0..15},
// lanes 16-31 hold K {16..31} (matches B fragment). Two aligned 16B loads.
__device__ __forceinline__ v16bf load_a_frag(const __bf16* __restrict__ p, int khalf) {
    union { v16bf v; v8bf h[2]; } u;
    u.h[0] = *(const v8bf*)(p + khalf * 16);
    u.h[1] = *(const v8bf*)(p + 8 + khalf * 16);
    return u.v;
}

__device__ __forceinline__ float sigmoid_f(float x) {
    return 1.0f / (1.0f + __expf(-x));
}

// ---------------------------------------------------------------------------
// Fused GRU step:  rp = h@U + b_rec ; xp = x@W + b_in  (gate order z,r,h)
//   z = sig(xz+rz); r = sig(xr+rr); hh = tanh(xh + r*rh); h' = z*h+(1-z)*hh
// Single kernel: GEMM over K = 512 (h part, feeds z/r/rh accums) then
// K = 128 (x part, feeds z/r/xh accums), then gate epilogue.
// UT layout: [1536][640] bf16, row n = [ U[:,n] (512) | W[:,n] (128) ].
// Grid (32,16), block 256 (8 waves): WG tile 128 rows x 32 gate-cols,
// wave tile 32x16 -> 8 v8f accumulators, 120 WMMA per wave.
// ---------------------------------------------------------------------------
__global__ __launch_bounds__(256)
void gru_step_kernel(const __bf16* __restrict__ h_bf,   // [B][H] bf16
                     const float*  __restrict__ h_f32,  // [B][H] f32
                     const __bf16* __restrict__ x_bf,   // [B][E] bf16 (time slice)
                     const __bf16* __restrict__ UT,     // [1536][640] bf16
                     const float*  __restrict__ bias,   // [2][1536] (b_in, b_rec)
                     float*  __restrict__ hn_f32,       // [B][H]
                     __bf16* __restrict__ hn_bf)        // [B][H]
{
    const int lane  = threadIdx.x & 31;
    const int wave  = threadIdx.x >> 5;
    const int mi    = wave & 3;        // 4 m-positions
    const int ni    = wave >> 2;       // 2 n-positions
    const int khalf = lane >> 4;
    const int l16   = lane & 15;

    const int rows0 = blockIdx.x * 128 + mi * 32;
    const int col   = blockIdx.y * 32 + ni * 16 + l16;   // H index of this lane

    const __bf16* __restrict__ Bz = UT + (size_t)col * KC_;
    const __bf16* __restrict__ Br = UT + (size_t)(col + H_) * KC_;
    const __bf16* __restrict__ Bh = UT + (size_t)(col + 2 * H_) * KC_;

    v8f accZ[2] = {}, accR[2] = {}, accRH[2] = {}, accXH[2] = {};

    // ---- recurrent part: K = 0..511 over h ----
    const size_t a0 = (size_t)(rows0 + l16) * H_;
    const size_t a1 = (size_t)(rows0 + 16 + l16) * H_;
    for (int kc = 0; kc < H_; kc += 32) {
        v16bf fa0 = load_a_frag(h_bf + a0 + kc, khalf);
        v16bf fa1 = load_a_frag(h_bf + a1 + kc, khalf);
        const int kb = kc + khalf * 16;
        v16bf fbz = *(const v16bf*)(Bz + kb);
        v16bf fbr = *(const v16bf*)(Br + kb);
        v16bf fbh = *(const v16bf*)(Bh + kb);
        accZ[0]  = wmma_bf16(fa0, fbz, accZ[0]);
        accZ[1]  = wmma_bf16(fa1, fbz, accZ[1]);
        accR[0]  = wmma_bf16(fa0, fbr, accR[0]);
        accR[1]  = wmma_bf16(fa1, fbr, accR[1]);
        accRH[0] = wmma_bf16(fa0, fbh, accRH[0]);
        accRH[1] = wmma_bf16(fa1, fbh, accRH[1]);
    }

    // ---- input part: K = 0..127 over x_t (weight rows 512..639 of UT) ----
    const size_t x0 = (size_t)(rows0 + l16) * E_;
    const size_t x1 = (size_t)(rows0 + 16 + l16) * E_;
    for (int kc = 0; kc < E_; kc += 32) {
        v16bf fa0 = load_a_frag(x_bf + x0 + kc, khalf);
        v16bf fa1 = load_a_frag(x_bf + x1 + kc, khalf);
        const int kb = H_ + kc + khalf * 16;
        v16bf fbz = *(const v16bf*)(Bz + kb);
        v16bf fbr = *(const v16bf*)(Br + kb);
        v16bf fbh = *(const v16bf*)(Bh + kb);
        accZ[0]  = wmma_bf16(fa0, fbz, accZ[0]);
        accZ[1]  = wmma_bf16(fa1, fbz, accZ[1]);
        accR[0]  = wmma_bf16(fa0, fbr, accR[0]);
        accR[1]  = wmma_bf16(fa1, fbr, accR[1]);
        accXH[0] = wmma_bf16(fa0, fbh, accXH[0]);
        accXH[1] = wmma_bf16(fa1, fbh, accXH[1]);
    }

    // ---- gate epilogue (C/D layout: N = lane%16, M = j + 8*(lane/16)) ----
    const float bZ  = bias[col]            + bias[G3_ + col];
    const float bR  = bias[H_ + col]       + bias[G3_ + H_ + col];
    const float bXH = bias[2 * H_ + col];
    const float bRH = bias[G3_ + 2 * H_ + col];

    #pragma unroll
    for (int mf = 0; mf < 2; ++mf) {
        #pragma unroll
        for (int j = 0; j < 8; ++j) {
            const int row = rows0 + mf * 16 + j + 8 * khalf;
            const size_t idx = (size_t)row * H_ + col;
            const float z  = sigmoid_f(accZ[mf][j] + bZ);
            const float r  = sigmoid_f(accR[mf][j] + bR);
            const float hh = tanhf(accXH[mf][j] + bXH + r * (accRH[mf][j] + bRH));
            const float ho = h_f32[idx];
            const float hn = z * ho + (1.0f - z) * hh;
            hn_f32[idx] = hn;
            hn_bf[idx]  = (__bf16)hn;
        }
    }
}

// ---------------------------------------------------------------------------
// Logits for one decoder step: out[b, t, :] = h_t[b, :] @ Wo + bo
// WoT: [VT][H] bf16 (column-major Wo). Grid (32, 8), block 256.
// ---------------------------------------------------------------------------
__global__ __launch_bounds__(256)
void logits_kernel(const __bf16* __restrict__ h_bf,   // [B][H] bf16
                   const __bf16* __restrict__ WoT,    // [VT][H] bf16
                   const float*  __restrict__ bo,     // [VT]
                   float* __restrict__ out,           // [B][T][VT]
                   int t)
{
    const int lane  = threadIdx.x & 31;
    const int wave  = threadIdx.x >> 5;
    const int mi    = wave & 3;
    const int ni    = wave >> 2;
    const int khalf = lane >> 4;
    const int l16   = lane & 15;

    const int rows0 = blockIdx.x * 128 + mi * 32;
    const int col   = blockIdx.y * 32 + ni * 16 + l16;   // vocab index

    const __bf16* __restrict__ Bv = WoT + (size_t)col * H_;
    const size_t a0 = (size_t)(rows0 + l16) * H_;
    const size_t a1 = (size_t)(rows0 + 16 + l16) * H_;

    v8f acc[2] = {};
    for (int kc = 0; kc < H_; kc += 32) {
        v16bf fa0 = load_a_frag(h_bf + a0 + kc, khalf);
        v16bf fa1 = load_a_frag(h_bf + a1 + kc, khalf);
        v16bf fb  = *(const v16bf*)(Bv + kc + khalf * 16);
        acc[0] = wmma_bf16(fa0, fb, acc[0]);
        acc[1] = wmma_bf16(fa1, fb, acc[1]);
    }

    const float bias = bo[col];
    #pragma unroll
    for (int mf = 0; mf < 2; ++mf) {
        #pragma unroll
        for (int j = 0; j < 8; ++j) {
            const int row = rows0 + mf * 16 + j + 8 * khalf;   // batch index
            out[(size_t)row * (T_ * VT_) + (size_t)t * VT_ + col] = acc[mf][j] + bias;
        }
    }
}

// ---------------------------------------------------------------------------
// Weight packing: UT[n][k] = (k<512 ? U[k][n] : W[k-512][n]) as bf16
// ---------------------------------------------------------------------------
__global__ void pack_gru_weights(const float* __restrict__ U,   // [512][1536]
                                 const float* __restrict__ W,   // [128][1536]
                                 __bf16* __restrict__ UT)       // [1536][640]
{
    const size_t i = (size_t)blockIdx.x * blockDim.x + threadIdx.x;  // < 1536*640
    const int k = (int)(i % KC_);
    const int n = (int)(i / KC_);
    const float v = (k < H_) ? U[(size_t)k * G3_ + n]
                             : W[(size_t)(k - H_) * G3_ + n];
    UT[i] = (__bf16)v;
}

__global__ void pack_out_weights(const float* __restrict__ Wo,  // [512][256]
                                 __bf16* __restrict__ WoT)      // [256][512]
{
    const size_t i = (size_t)blockIdx.x * blockDim.x + threadIdx.x;  // < 256*512
    const int k = (int)(i % H_);
    const int n = (int)(i / H_);
    WoT[i] = (__bf16)Wo[(size_t)k * VT_ + n];
}

// ---------------------------------------------------------------------------
// Embedding gathers (to time-major bf16 activation matrices)
// ---------------------------------------------------------------------------
__global__ void gather_src_kernel(const int* __restrict__ source,   // [B][S]
                                  const float* __restrict__ emb,    // [VS][E]
                                  __bf16* __restrict__ xout)        // [S][B][E]
{
    const size_t i = (size_t)blockIdx.x * blockDim.x + threadIdx.x;  // < S*B*E
    const int e = (int)(i & (E_ - 1));
    const size_t sb = i >> 7;
    const int b = (int)(sb & (B_ - 1));
    const int s = (int)(sb >> 12);
    const int tok = source[(size_t)b * S_ + s];
    xout[i] = (__bf16)emb[(size_t)tok * E_ + e];
}

__global__ void gather_dec_kernel(const int* __restrict__ targets,  // [B][T]
                                  const float* __restrict__ emb,    // [VT][E]
                                  __bf16* __restrict__ xout)        // [T][B][E]
{
    const size_t i = (size_t)blockIdx.x * blockDim.x + threadIdx.x;  // < T*B*E
    const int e = (int)(i & (E_ - 1));
    const size_t tb = i >> 7;
    const int b = (int)(tb & (B_ - 1));
    const int t = (int)(tb >> 12);
    const int tok = (t == 0) ? 1 /*BOW*/ : targets[(size_t)b * T_ + (t - 1)];
    xout[i] = (__bf16)emb[(size_t)tok * E_ + e];
}

// decoder initial state = fwd final + bwd final (f32 + bf16 views)
__global__ void sum_states_kernel(const float* __restrict__ a,
                                  const float* __restrict__ b,
                                  float* __restrict__ of,
                                  __bf16* __restrict__ ob)
{
    const size_t i = (size_t)blockIdx.x * blockDim.x + threadIdx.x;  // < B*H
    const float v = a[i] + b[i];
    of[i] = v;
    ob[i] = (__bf16)v;
}

// ---------------------------------------------------------------------------
extern "C" void kernel_launch(void* const* d_in, const int* in_sizes, int n_in,
                              void* d_out, int out_size, void* d_ws, size_t ws_size,
                              hipStream_t stream)
{
    (void)in_sizes; (void)n_in; (void)out_size; (void)ws_size;

    const int*   source  = (const int*)d_in[0];
    const int*   targets = (const int*)d_in[1];
    const float* src_emb = (const float*)d_in[2];
    const float* tgt_emb = (const float*)d_in[3];
    const float* Wf = (const float*)d_in[4];
    const float* Uf = (const float*)d_in[5];
    const float* bf = (const float*)d_in[6];
    const float* Wb = (const float*)d_in[7];
    const float* Ub = (const float*)d_in[8];
    const float* bb = (const float*)d_in[9];
    const float* Wd = (const float*)d_in[10];
    const float* Ud = (const float*)d_in[11];
    const float* bd = (const float*)d_in[12];
    const float* Wo = (const float*)d_in[13];
    const float* bo = (const float*)d_in[14];
    float* out = (float*)d_out;

    // ---- workspace carve-out (256B aligned regions) ----
    char* ws = (char*)d_ws;
    size_t off = 0;
    auto take = [&](size_t bytes) -> char* {
        char* p = ws + off;
        off = (off + bytes + 255) & ~(size_t)255;
        return p;
    };
    __bf16* UTf  = (__bf16*)take((size_t)G3_ * KC_ * 2);
    __bf16* UTb  = (__bf16*)take((size_t)G3_ * KC_ * 2);
    __bf16* UTd  = (__bf16*)take((size_t)G3_ * KC_ * 2);
    __bf16* WoT  = (__bf16*)take((size_t)VT_ * H_ * 2);
    __bf16* xsrc = (__bf16*)take((size_t)S_ * B_ * E_ * 2);
    __bf16* xdec = (__bf16*)take((size_t)T_ * B_ * E_ * 2);
    float*  hf32[4];
    __bf16* hbf[4];
    for (int i = 0; i < 4; ++i) hf32[i] = (float*)take((size_t)B_ * H_ * 4);
    for (int i = 0; i < 4; ++i) hbf[i]  = (__bf16*)take((size_t)B_ * H_ * 2);

    // zero initial encoder states (fwd uses buf0, bwd uses buf2)
    hipMemsetAsync(hf32[0], 0, (size_t)B_ * H_ * 4, stream);
    hipMemsetAsync(hbf[0],  0, (size_t)B_ * H_ * 2, stream);
    hipMemsetAsync(hf32[2], 0, (size_t)B_ * H_ * 4, stream);
    hipMemsetAsync(hbf[2],  0, (size_t)B_ * H_ * 2, stream);

    // weight packing + embedding gathers
    pack_gru_weights<<<(G3_ * KC_) / 256, 256, 0, stream>>>(Uf, Wf, UTf);
    pack_gru_weights<<<(G3_ * KC_) / 256, 256, 0, stream>>>(Ub, Wb, UTb);
    pack_gru_weights<<<(G3_ * KC_) / 256, 256, 0, stream>>>(Ud, Wd, UTd);
    pack_out_weights<<<(VT_ * H_) / 256, 256, 0, stream>>>(Wo, WoT);
    gather_src_kernel<<<(int)(((size_t)S_ * B_ * E_) / 256), 256, 0, stream>>>(source, src_emb, xsrc);
    gather_dec_kernel<<<(int)(((size_t)T_ * B_ * E_) / 256), 256, 0, stream>>>(targets, tgt_emb, xdec);

    const dim3 gs(B_ / 128, H_ / 32);    // 32 x 16 workgroups per GRU step
    const dim3 gl(B_ / 128, VT_ / 32);   // 32 x 8 workgroups per logits step

    // ---- forward encoder (ping-pong buffers 0/1; 24 steps -> final in 0) ----
    int cur = 0;
    for (int s = 0; s < S_; ++s) {
        const int nxt = 1 - cur;
        gru_step_kernel<<<gs, 256, 0, stream>>>(hbf[cur], hf32[cur],
            xsrc + (size_t)s * B_ * E_, UTf, bf, hf32[nxt], hbf[nxt]);
        cur = nxt;
    }

    // ---- backward encoder (buffers 2/3; reversed time; final in 2) ----
    int curb = 2;
    for (int i = 0; i < S_; ++i) {
        const int nxt = (curb == 2) ? 3 : 2;
        gru_step_kernel<<<gs, 256, 0, stream>>>(hbf[curb], hf32[curb],
            xsrc + (size_t)(S_ - 1 - i) * B_ * E_, UTb, bb, hf32[nxt], hbf[nxt]);
        curb = nxt;
    }

    // ---- decoder init: buf1 = fwd_final + bwd_final ----
    sum_states_kernel<<<(B_ * H_) / 256, 256, 0, stream>>>(hf32[0], hf32[2], hf32[1], hbf[1]);

    // ---- decoder (buffers 1/3), logits after every step ----
    int curd = 1;
    for (int t = 0; t < T_; ++t) {
        const int nxt = (curd == 1) ? 3 : 1;
        gru_step_kernel<<<gs, 256, 0, stream>>>(hbf[curd], hf32[curd],
            xdec + (size_t)t * B_ * E_, UTd, bd, hf32[nxt], hbf[nxt]);
        logits_kernel<<<gl, 256, 0, stream>>>(hbf[nxt], WoT, bo, out, t);
        curd = nxt;
    }
}
